// MoE_833223655783
// MI455X (gfx1250) — compile-verified, matrix operand order checked by reading, not run on hardware
//
#include <hip/hip_runtime.h>
#include <hip/hip_bf16.h>

// Problem constants (from reference)
#define TOK   8192      // B*S
#define DIM_  1024
#define HID_  1280
#define NEXP  8
#define NSH   2
#define NTOT  10        // routed + shared unified expert loop
#define CWS   16        // combine-weight row stride (padded)

typedef __attribute__((ext_vector_type(16))) __bf16 bf16x16;
typedef __attribute__((ext_vector_type(8)))  float  f32x8;

// ---------------------------------------------------------------------------
// WMMA helpers
// ---------------------------------------------------------------------------
__device__ __forceinline__ f32x8 wmma_bf16(bf16x16 a, bf16x16 b, f32x8 c) {
    // 8 args: (neg_a, A, neg_b, B, c_mod, C, reuse_a, reuse_b)
    return __builtin_amdgcn_wmma_f32_16x16x32_bf16(false, a, false, b,
                                                   (short)0, c, false, false);
}

// CDNA5 16-bit A/B fragment layout: lane = (row%16) + 16*h, elements 0..7 are
// K = k0+8h..k0+8h+7, elements 8..15 are K = k0+16+8h..+7  -> two b128 loads
// from a row-major bf16 matrix.
__device__ __forceinline__ bf16x16 load_frag(const __hip_bfloat16* rowPtr,
                                             int k0, int h) {
    union { bf16x16 v; int4 q[2]; } u;
    const char* p = (const char*)(rowPtr + k0 + 8 * h);
    u.q[0] = *(const int4*)(p);
    u.q[1] = *(const int4*)(p + 32);
    return u.v;
}

__device__ __forceinline__ float silu(float s) {
    return s / (1.0f + __expf(-s));
}

// ---------------------------------------------------------------------------
// Zero-init output (d_out is poisoned by harness)
// ---------------------------------------------------------------------------
__global__ void init_out_kernel(float* __restrict__ out, long n) {
    long i = (long)blockIdx.x * blockDim.x + threadIdx.x;
    long stride = (long)gridDim.x * blockDim.x;
    for (; i < n; i += stride) out[i] = 0.0f;
}

// ---------------------------------------------------------------------------
// fp32 -> bf16 cast (vectorized x4)
// ---------------------------------------------------------------------------
__global__ void cast_bf16_kernel(const float* __restrict__ src,
                                 __hip_bfloat16* __restrict__ dst, long n4) {
    long i = (long)blockIdx.x * blockDim.x + threadIdx.x;
    long stride = (long)gridDim.x * blockDim.x;
    for (; i < n4; i += stride) {
        float4 f = ((const float4*)src)[i];
        union { __hip_bfloat16 h[4]; uint2 u; } o;
        o.h[0] = __float2bfloat16(f.x);
        o.h[1] = __float2bfloat16(f.y);
        o.h[2] = __float2bfloat16(f.z);
        o.h[3] = __float2bfloat16(f.w);
        ((uint2*)dst)[i] = o.u;
    }
}

// ---------------------------------------------------------------------------
// Gate: sigmoid scores + bias, top-2, normalized weights.
// cw[t*CWS + e] = combine weight (0 if not routed); slots 8,9 = 1/NSH for the
// shared experts. Slot 10 holds the per-token entropy contribution.
// One wave32 per token.
// ---------------------------------------------------------------------------
__global__ void gate_kernel(const float* __restrict__ x,
                            const float* __restrict__ emb,
                            const float* __restrict__ gbias,
                            float* __restrict__ cw) {
    int wid  = ((int)blockIdx.x * blockDim.x + threadIdx.x) >> 5;
    int lane = threadIdx.x & 31;
    if (wid >= TOK) return;
    const float* xr = x + (size_t)wid * DIM_;
    float p[NEXP];
#pragma unroll
    for (int e = 0; e < NEXP; ++e) p[e] = 0.0f;
    for (int k = lane; k < DIM_; k += 32) {
        float xv = xr[k];
#pragma unroll
        for (int e = 0; e < NEXP; ++e) p[e] += xv * emb[e * DIM_ + k];
    }
#pragma unroll
    for (int e = 0; e < NEXP; ++e)
        for (int off = 16; off > 0; off >>= 1)
            p[e] += __shfl_xor(p[e], off, 32);
    if (lane == 0) {
        float s[NEXP];
#pragma unroll
        for (int e = 0; e < NEXP; ++e)
            s[e] = 1.0f / (1.0f + __expf(-p[e])) + gbias[e];
        int i0 = 0;
#pragma unroll
        for (int e = 1; e < NEXP; ++e) if (s[e] > s[i0]) i0 = e;   // first max
        int i1 = (i0 == 0) ? 1 : 0;
#pragma unroll
        for (int e = 0; e < NEXP; ++e)
            if (e != i0 && s[e] > s[i1]) i1 = e;
        float v0 = s[i0], v1 = s[i1];
        float inv = 1.0f / (v0 + v1);
        float w0 = v0 * inv, w1 = v1 * inv;
        float* row = cw + (size_t)wid * CWS;
#pragma unroll
        for (int e = 0; e < NEXP; ++e)
            row[e] = (e == i0) ? w0 : ((e == i1) ? w1 : 0.0f);
        row[8] = 1.0f / (float)NSH;   // shared experts: mean
        row[9] = 1.0f / (float)NSH;
        row[10] = -(w0 * __logf(w0) + w1 * __logf(w1));
    }
}

// Deterministic entropy reduction: one block, fixed accumulation order.
__global__ void entropy_reduce_kernel(const float* __restrict__ cw,
                                      float* __restrict__ ent_out) {
    __shared__ float buf[256];
    float acc = 0.0f;
    for (int t = threadIdx.x; t < TOK; t += 256)
        acc += cw[(size_t)t * CWS + 10];
    buf[threadIdx.x] = acc;
    __syncthreads();
    for (int off = 128; off > 0; off >>= 1) {
        if (threadIdx.x < off) buf[threadIdx.x] += buf[threadIdx.x + off];
        __syncthreads();
    }
    if (threadIdx.x == 0) *ent_out = buf[0] / (float)TOK;
}

// ---------------------------------------------------------------------------
// Pass 1: H = silu(x @ W1^T) * (x @ W3^T)   [M=TOK, N=HID, K=DIM], bf16 out.
// Block = 8 waves = 128x64 tile. Wave = 32x32 patch for BOTH GEMMs
// (A fragments reused across 4 B fragments). Software-pipelined: fragments
// for k0+32 are issued before the WMMAs for k0 so loads overlap matrix ops.
// ---------------------------------------------------------------------------
__global__ __launch_bounds__(256)
void ffn_gate_up_kernel(const __hip_bfloat16* __restrict__ xb,
                        const __hip_bfloat16* __restrict__ w1e,
                        const __hip_bfloat16* __restrict__ w3e,
                        __hip_bfloat16* __restrict__ Hd) {
    const int lane = threadIdx.x & 31;
    const int wave = threadIdx.x >> 5;
    const int wm = wave & 3, wn = wave >> 2;
    const int m0 = blockIdx.y * 128 + wm * 32;
    const int n0 = blockIdx.x * 64 + wn * 32;
    const int r = lane & 15, h = lane >> 4;

    const __hip_bfloat16* aR0 = xb  + (size_t)(m0 + r) * DIM_;
    const __hip_bfloat16* aR1 = xb  + (size_t)(m0 + 16 + r) * DIM_;
    const __hip_bfloat16* bR0 = w1e + (size_t)(n0 + r) * DIM_;
    const __hip_bfloat16* bR1 = w1e + (size_t)(n0 + 16 + r) * DIM_;
    const __hip_bfloat16* bR2 = w3e + (size_t)(n0 + r) * DIM_;
    const __hip_bfloat16* bR3 = w3e + (size_t)(n0 + 16 + r) * DIM_;

    f32x8 acc[8];
#pragma unroll
    for (int i = 0; i < 8; ++i) acc[i] = (f32x8){};

    // prologue: k0 = 0 fragments
    bf16x16 a0 = load_frag(aR0, 0, h);
    bf16x16 a1 = load_frag(aR1, 0, h);
    bf16x16 b0 = load_frag(bR0, 0, h);
    bf16x16 b1 = load_frag(bR1, 0, h);
    bf16x16 b2 = load_frag(bR2, 0, h);
    bf16x16 b3 = load_frag(bR3, 0, h);

    for (int k0 = 0; k0 < DIM_ - 32; k0 += 32) {
        const int kn = k0 + 32;
        // prefetch streamed weight rows two K-steps ahead (global_prefetch_b8)
        __builtin_prefetch(bR0 + kn + 32, 0, 1);
        __builtin_prefetch(bR1 + kn + 32, 0, 1);
        __builtin_prefetch(bR2 + kn + 32, 0, 1);
        __builtin_prefetch(bR3 + kn + 32, 0, 1);
        // issue next-step loads first so they overlap this step's WMMAs
        bf16x16 na0 = load_frag(aR0, kn, h);
        bf16x16 na1 = load_frag(aR1, kn, h);
        bf16x16 nb0 = load_frag(bR0, kn, h);
        bf16x16 nb1 = load_frag(bR1, kn, h);
        bf16x16 nb2 = load_frag(bR2, kn, h);
        bf16x16 nb3 = load_frag(bR3, kn, h);

        acc[0] = wmma_bf16(a0, b0, acc[0]);
        acc[1] = wmma_bf16(a0, b1, acc[1]);
        acc[2] = wmma_bf16(a0, b2, acc[2]);
        acc[3] = wmma_bf16(a0, b3, acc[3]);
        acc[4] = wmma_bf16(a1, b0, acc[4]);
        acc[5] = wmma_bf16(a1, b1, acc[5]);
        acc[6] = wmma_bf16(a1, b2, acc[6]);
        acc[7] = wmma_bf16(a1, b3, acc[7]);

        a0 = na0; a1 = na1; b0 = nb0; b1 = nb1; b2 = nb2; b3 = nb3;
    }
    acc[0] = wmma_bf16(a0, b0, acc[0]);
    acc[1] = wmma_bf16(a0, b1, acc[1]);
    acc[2] = wmma_bf16(a0, b2, acc[2]);
    acc[3] = wmma_bf16(a0, b3, acc[3]);
    acc[4] = wmma_bf16(a1, b0, acc[4]);
    acc[5] = wmma_bf16(a1, b1, acc[5]);
    acc[6] = wmma_bf16(a1, b2, acc[6]);
    acc[7] = wmma_bf16(a1, b3, acc[7]);

    // D layout: lane l covers col n0(+16) + (l&15), rows mbase + 8*(l>>4) + v
    const int col = n0 + r;
#pragma unroll
    for (int am = 0; am < 2; ++am) {
        const int row0 = m0 + 16 * am + 8 * h;
        const f32x8 g0 = acc[am * 4 + 0];   // wait, indexing below
        (void)g0;
#pragma unroll
        for (int v = 0; v < 8; ++v) {
            float hv0 = silu(acc[am * 4 + 0][v]) * acc[am * 4 + 2][v];
            float hv1 = silu(acc[am * 4 + 1][v]) * acc[am * 4 + 3][v];
            Hd[(size_t)(row0 + v) * HID_ + col]      = __float2bfloat16(hv0);
            Hd[(size_t)(row0 + v) * HID_ + col + 16] = __float2bfloat16(hv1);
        }
    }
}

// ---------------------------------------------------------------------------
// Pass 2: out += cw[t,e] * (H @ W2^T)   [M=TOK, N=DIM, K=HID], f32 RMW.
// Block = 8 waves = 128x64 tile; wave = 32x32 patch. Software-pipelined.
// Expert kernels are stream-ordered; each block owns a unique out tile.
// ---------------------------------------------------------------------------
__global__ __launch_bounds__(256)
void ffn_down_kernel(const __hip_bfloat16* __restrict__ Hd,
                     const __hip_bfloat16* __restrict__ w2e,
                     const float* __restrict__ cw, int eIdx,
                     float* __restrict__ out) {
    __shared__ float cwS[128];
    const int lane = threadIdx.x & 31;
    const int wave = threadIdx.x >> 5;
    const int wm = wave & 3, wn = wave >> 2;
    const int mBlk = blockIdx.y * 128;
    const int m0 = mBlk + wm * 32;
    const int n0 = blockIdx.x * 64 + wn * 32;
    const int r = lane & 15, h = lane >> 4;

    if (threadIdx.x < 128)
        cwS[threadIdx.x] = cw[(size_t)(mBlk + threadIdx.x) * CWS + eIdx];
    __syncthreads();

    const __hip_bfloat16* aR0 = Hd  + (size_t)(m0 + r) * HID_;
    const __hip_bfloat16* aR1 = Hd  + (size_t)(m0 + 16 + r) * HID_;
    const __hip_bfloat16* bR0 = w2e + (size_t)(n0 + r) * HID_;
    const __hip_bfloat16* bR1 = w2e + (size_t)(n0 + 16 + r) * HID_;

    f32x8 acc[4];
#pragma unroll
    for (int i = 0; i < 4; ++i) acc[i] = (f32x8){};

    bf16x16 a0 = load_frag(aR0, 0, h);
    bf16x16 a1 = load_frag(aR1, 0, h);
    bf16x16 b0 = load_frag(bR0, 0, h);
    bf16x16 b1 = load_frag(bR1, 0, h);

    for (int k0 = 0; k0 < HID_ - 32; k0 += 32) {
        const int kn = k0 + 32;
        __builtin_prefetch(bR0 + kn + 32, 0, 1);
        __builtin_prefetch(bR1 + kn + 32, 0, 1);
        bf16x16 na0 = load_frag(aR0, kn, h);
        bf16x16 na1 = load_frag(aR1, kn, h);
        bf16x16 nb0 = load_frag(bR0, kn, h);
        bf16x16 nb1 = load_frag(bR1, kn, h);

        acc[0] = wmma_bf16(a0, b0, acc[0]);
        acc[1] = wmma_bf16(a0, b1, acc[1]);
        acc[2] = wmma_bf16(a1, b0, acc[2]);
        acc[3] = wmma_bf16(a1, b1, acc[3]);

        a0 = na0; a1 = na1; b0 = nb0; b1 = nb1;
    }
    acc[0] = wmma_bf16(a0, b0, acc[0]);
    acc[1] = wmma_bf16(a0, b1, acc[1]);
    acc[2] = wmma_bf16(a1, b0, acc[2]);
    acc[3] = wmma_bf16(a1, b1, acc[3]);

    const int col = n0 + r;
#pragma unroll
    for (int am = 0; am < 2; ++am) {
        const int row0 = m0 + 16 * am + 8 * h;
#pragma unroll
        for (int v = 0; v < 8; ++v) {
            float cww = cwS[wm * 32 + 16 * am + 8 * h + v];
            float* o = out + (size_t)(row0 + v) * DIM_ + col;
            o[0]  += cww * acc[am * 2 + 0][v];
            o[16] += cww * acc[am * 2 + 1][v];
        }
    }
}

// ---------------------------------------------------------------------------
// Host launcher
// ---------------------------------------------------------------------------
extern "C" void kernel_launch(void* const* d_in, const int* in_sizes, int n_in,
                              void* d_out, int out_size, void* d_ws, size_t ws_size,
                              hipStream_t stream) {
    const float* x    = (const float*)d_in[0];
    const float* emb  = (const float*)d_in[1];
    const float* gbia = (const float*)d_in[2];
    const float* w1   = (const float*)d_in[3];
    const float* w2   = (const float*)d_in[4];
    const float* w3   = (const float*)d_in[5];
    const float* sw1  = (const float*)d_in[6];
    const float* sw2  = (const float*)d_in[7];
    const float* sw3  = (const float*)d_in[8];
    float* out = (float*)d_out;

    // Workspace layout (bytes, 256-aligned):
    const long nX  = (long)TOK * DIM_;           // 8,388,608
    const long nW  = (long)NEXP * HID_ * DIM_;   // 10,485,760
    const long nSW = (long)NSH  * HID_ * DIM_;   //  2,621,440
    const long nH  = (long)TOK * HID_;           // 10,485,760

    char* ws = (char*)d_ws;
    size_t off = 0;
    float* cwBuf = (float*)(ws + off);                  off += (size_t)TOK * CWS * 4;
    __hip_bfloat16* xb   = (__hip_bfloat16*)(ws + off); off += (size_t)nX  * 2;
    __hip_bfloat16* w1b  = (__hip_bfloat16*)(ws + off); off += (size_t)nW  * 2;
    __hip_bfloat16* w3b  = (__hip_bfloat16*)(ws + off); off += (size_t)nW  * 2;
    __hip_bfloat16* w2b  = (__hip_bfloat16*)(ws + off); off += (size_t)nW  * 2;
    __hip_bfloat16* sw1b = (__hip_bfloat16*)(ws + off); off += (size_t)nSW * 2;
    __hip_bfloat16* sw3b = (__hip_bfloat16*)(ws + off); off += (size_t)nSW * 2;
    __hip_bfloat16* sw2b = (__hip_bfloat16*)(ws + off); off += (size_t)nSW * 2;
    __hip_bfloat16* Hbuf = (__hip_bfloat16*)(ws + off); off += (size_t)nH  * 2;
    // total ~117 MB

    // 1) zero output (incl. aux_loss / entropy slots)
    init_out_kernel<<<8192, 256, 0, stream>>>(out, (long)out_size);

    // 2) casts to bf16
    cast_bf16_kernel<<<4096, 256, 0, stream>>>(x,   xb,   nX  / 4);
    cast_bf16_kernel<<<8192, 256, 0, stream>>>(w1,  w1b,  nW  / 4);
    cast_bf16_kernel<<<8192, 256, 0, stream>>>(w3,  w3b,  nW  / 4);
    cast_bf16_kernel<<<8192, 256, 0, stream>>>(w2,  w2b,  nW  / 4);
    cast_bf16_kernel<<<2048, 256, 0, stream>>>(sw1, sw1b, nSW / 4);
    cast_bf16_kernel<<<2048, 256, 0, stream>>>(sw3, sw3b, nSW / 4);
    cast_bf16_kernel<<<2048, 256, 0, stream>>>(sw2, sw2b, nSW / 4);

    // 3) gate + deterministic entropy
    gate_kernel<<<TOK * 32 / 256, 256, 0, stream>>>(x, emb, gbia, cwBuf);
    entropy_reduce_kernel<<<1, 256, 0, stream>>>(cwBuf, out + (size_t)TOK * DIM_ + 1);
    // aux_loss slot (out + TOK*DIM) stays 0 from init.

    // 4) unified expert loop: 8 routed (combine weights) + 2 shared (weight 1/NSH)
    const long perW = (long)HID_ * DIM_;
    dim3 g1(HID_ / 64, TOK / 128);  // 20 x 64
    dim3 g2(DIM_ / 64, TOK / 128);  // 16 x 64
    for (int e = 0; e < NTOT; ++e) {
        const __hip_bfloat16* pw1 = (e < NEXP) ? (w1b + (size_t)e * perW)
                                               : (sw1b + (size_t)(e - NEXP) * perW);
        const __hip_bfloat16* pw3 = (e < NEXP) ? (w3b + (size_t)e * perW)
                                               : (sw3b + (size_t)(e - NEXP) * perW);
        const __hip_bfloat16* pw2 = (e < NEXP) ? (w2b + (size_t)e * perW)
                                               : (sw2b + (size_t)(e - NEXP) * perW);
        ffn_gate_up_kernel<<<g1, 256, 0, stream>>>(xb, pw1, pw3, Hbuf);
        ffn_down_kernel<<<g2, 256, 0, stream>>>(Hbuf, pw2, cwBuf, e, out);
    }
    (void)in_sizes; (void)n_in; (void)ws_size;
}
